// AutoEncoder_14207751815410
// MI455X (gfx1250) — matrix operand order repaired and checked by reading, compiled, not run on hardware
//
#include <hip/hip_runtime.h>

#define B_DIM 4096
#define E_DIM 1024
#define F_DIM 32768
#define TOPK 64
#define DTOPK 512
#define DEAD_CUTOFF 50000

typedef __attribute__((ext_vector_type(16))) __bf16 v16bf;
typedef __attribute__((ext_vector_type(8)))  __bf16 v8bf;
typedef __attribute__((ext_vector_type(8)))  float  v8f;

union FragBF { v16bf v; struct { v8bf lo, hi; } h; };

#define LSTRIDE 40  // bf16 units per LDS row: 80B -> 16B aligned, conflict-free b128 reads

// ---------------------------------------------------------------------------
// Kernel 1: project[b, f] = sum_e (embed[b,e] - bias[e]) * W[f,e]
// bf16x3 split-precision WMMA GEMM; block tile 64(M) x 64(N), K staged 32-wide.
// 8 waves; wave w owns m-tile (w&3) and n-tiles {2*(w>>2), 2*(w>>2)+1}.
// Staging is fully vectorized: float4 global loads -> packed v8bf LDS stores.
// ---------------------------------------------------------------------------
__global__ __launch_bounds__(256) void gemm_proj_kernel(
    const float* __restrict__ embed, const float* __restrict__ enc_bias,
    const float* __restrict__ encW, float* __restrict__ proj, int r0)
{
  __shared__ alignas(16) __bf16 sAh[64 * LSTRIDE];
  __shared__ alignas(16) __bf16 sAl[64 * LSTRIDE];
  __shared__ alignas(16) __bf16 sBh[64 * LSTRIDE];
  __shared__ alignas(16) __bf16 sBl[64 * LSTRIDE];

  const int t    = threadIdx.x;
  const int lane = t & 31;
  const int wv   = t >> 5;        // wave 0..7
  const int mt   = wv & 3;        // m-tile 0..3
  const int nt0  = (wv >> 2) * 2; // n-tiles nt0, nt0+1

  const int b0 = r0 + blockIdx.x * 64;  // global batch row base
  const int f0 = blockIdx.y * 64;       // feature base

  v8f acc0 = {};
  v8f acc1 = {};

  // ---- staging assignment: thread t owns row sr = t/4, cols sc..sc+7 ----
  const int sr = t >> 2;
  const int sc = (t & 3) * 8;
  const float* eptr = &embed[(size_t)(b0 + sr) * E_DIM + sc];
  const float* wptr = &encW [(size_t)(f0 + sr) * E_DIM + sc];
  const float* bptr = &enc_bias[sc];
  __bf16* sAhp = &sAh[sr * LSTRIDE + sc];
  __bf16* sAlp = &sAl[sr * LSTRIDE + sc];
  __bf16* sBhp = &sBh[sr * LSTRIDE + sc];
  __bf16* sBlp = &sBl[sr * LSTRIDE + sc];

  // ---- fragment addressing ----
  // A (16-bit 16x32): lanes 0-15 M=lane K-base 0; lanes 16-31 M=lane-16 K-base +8;
  // elements 0..7 -> K c..c+7, elements 8..15 -> K c+16..c+23
  const int arow = mt * 16 + (lane & 15);
  const int aoff = arow * LSTRIDE + ((lane >> 4) * 8);
  // B (16-bit 32x16): lanes 0-15 N=lane K-base 0; lanes 16-31 N=lane-16 K-base +8; same K order as A
  const int brow0 = nt0 * 16 + (lane & 15);
  const int brow1 = brow0 + 16;
  const int bkb   = (lane >> 4) * 8;
  const int boff0 = brow0 * LSTRIDE + bkb;
  const int boff1 = brow1 * LSTRIDE + bkb;

  for (int kc = 0; kc < E_DIM; kc += 32) {
    // ---- vectorized cooperative stage (straight-line, no inner loop) ----
    {
      const float4 ev0 = *(const float4*)(eptr + kc);
      const float4 ev1 = *(const float4*)(eptr + kc + 4);
      const float4 bv0 = *(const float4*)(bptr + kc);
      const float4 bv1 = *(const float4*)(bptr + kc + 4);
      const float4 wv0 = *(const float4*)(wptr + kc);
      const float4 wv1 = *(const float4*)(wptr + kc + 4);
      const float av[8] = { ev0.x - bv0.x, ev0.y - bv0.y, ev0.z - bv0.z, ev0.w - bv0.w,
                            ev1.x - bv1.x, ev1.y - bv1.y, ev1.z - bv1.z, ev1.w - bv1.w };
      const float wvv[8] = { wv0.x, wv0.y, wv0.z, wv0.w, wv1.x, wv1.y, wv1.z, wv1.w };
      v8bf ahv, alv, whv, wlv;
#pragma unroll
      for (int j = 0; j < 8; ++j) {
        const __bf16 h = (__bf16)av[j];
        ahv[j] = h;
        alv[j] = (__bf16)(av[j] - (float)h);
        const __bf16 g = (__bf16)wvv[j];
        whv[j] = g;
        wlv[j] = (__bf16)(wvv[j] - (float)g);
      }
      *(v8bf*)sAhp = ahv;
      *(v8bf*)sAlp = alv;
      *(v8bf*)sBhp = whv;
      *(v8bf*)sBlp = wlv;
    }
    __syncthreads();

    FragBF ah, al, bh0, bl0, bh1, bl1;
    ah.h.lo  = *(const v8bf*)&sAh[aoff];
    ah.h.hi  = *(const v8bf*)&sAh[aoff + 16];
    al.h.lo  = *(const v8bf*)&sAl[aoff];
    al.h.hi  = *(const v8bf*)&sAl[aoff + 16];
    bh0.h.lo = *(const v8bf*)&sBh[boff0];
    bh0.h.hi = *(const v8bf*)&sBh[boff0 + 16];
    bl0.h.lo = *(const v8bf*)&sBl[boff0];
    bl0.h.hi = *(const v8bf*)&sBl[boff0 + 16];
    bh1.h.lo = *(const v8bf*)&sBh[boff1];
    bh1.h.hi = *(const v8bf*)&sBh[boff1 + 16];
    bl1.h.lo = *(const v8bf*)&sBl[boff1];
    bl1.h.hi = *(const v8bf*)&sBl[boff1 + 16];

    // bf16x3: hi*hi + hi*lo + lo*hi  (~fp32 product accuracy)
    acc0 = __builtin_amdgcn_wmma_f32_16x16x32_bf16(false, ah.v, false, bh0.v, (short)0, acc0, false, false);
    acc0 = __builtin_amdgcn_wmma_f32_16x16x32_bf16(false, ah.v, false, bl0.v, (short)0, acc0, false, false);
    acc0 = __builtin_amdgcn_wmma_f32_16x16x32_bf16(false, al.v, false, bh0.v, (short)0, acc0, false, false);
    acc1 = __builtin_amdgcn_wmma_f32_16x16x32_bf16(false, ah.v, false, bh1.v, (short)0, acc1, false, false);
    acc1 = __builtin_amdgcn_wmma_f32_16x16x32_bf16(false, ah.v, false, bl1.v, (short)0, acc1, false, false);
    acc1 = __builtin_amdgcn_wmma_f32_16x16x32_bf16(false, al.v, false, bh1.v, (short)0, acc1, false, false);
    __syncthreads();
  }

  // Epilogue: C layout -> VGPR r: M = r (+8 for lanes 16-31), N = lane&15
  const int orow = blockIdx.x * 64 + mt * 16 + ((lane >> 4) * 8);  // chunk-local row
  const int col0 = f0 + nt0 * 16 + (lane & 15);
#pragma unroll
  for (int r2 = 0; r2 < 8; ++r2) {
    proj[(size_t)(orow + r2) * F_DIM + col0]      = acc0[r2];
    proj[(size_t)(orow + r2) * F_DIM + col0 + 16] = acc1[r2];
  }
}

// ---------------------------------------------------------------------------
// Kernel 2: per-row exact top-K (radix select over monotone float keys),
// deterministic compaction, then weighted gather reconstruction.
// pass 0: top-64 over all features -> embed_recon (+bias)
// pass 1: top-512 over dead-masked features -> dead_recon
// Dead mask cached once per block as an LDS bitmap built with wave32 ballots.
// ---------------------------------------------------------------------------
__device__ inline unsigned int mapkey(float x) {
  unsigned int u = __float_as_uint(x);
  return (u & 0x80000000u) ? ~u : (u | 0x80000000u);  // monotone: bigger float -> bigger key
}

__global__ __launch_bounds__(256) void select_recon_kernel(
    const float* __restrict__ proj, const float* __restrict__ lookup,
    const float* __restrict__ enc_bias, const int* __restrict__ last_usage,
    float* __restrict__ out, int r0)
{
  const int t = threadIdx.x;
  const int b = r0 + blockIdx.x;                  // global batch row
  const float* row = proj + (size_t)blockIdx.x * F_DIM;

  __shared__ unsigned int hist[256];
  __shared__ unsigned int cntGt[256], cntEq[256];
  __shared__ int   selF[DTOPK];
  __shared__ float selW[DTOPK];
  __shared__ unsigned int deadmask[F_DIM / 32];   // 4 KB bitmap
  __shared__ unsigned int s_prefix, s_needed;

  // Build dead-feature bitmap once: one ballot per 32 features.
  {
    const int wv = t >> 5, lane = t & 31;
    for (int m = wv; m < F_DIM / 32; m += 8) {
      const unsigned long long bal = __ballot(last_usage[m * 32 + lane] > DEAD_CUTOFF);
      if (lane == 0) deadmask[m] = (unsigned int)bal;
    }
  }
  __syncthreads();
  const int mysh = t & 31;  // strided i keeps (i & 31) == (t & 31)

  for (int pass = 0; pass < 2; ++pass) {
    const int K = (pass == 0) ? TOPK : DTOPK;
    const bool masked = (pass != 0);

    if (t == 0) { s_prefix = 0u; s_needed = (unsigned)K; }
    __syncthreads();

    // 4-round radix select: exact 32-bit threshold key T and tie count
    for (int round = 0; round < 4; ++round) {
      hist[t] = 0u;
      __syncthreads();
      const unsigned int prefix = s_prefix;
      const int shift = 24 - round * 8;
      for (int i = t; i < F_DIM; i += 256) {
        if (masked && !((deadmask[i >> 5] >> mysh) & 1u)) continue;
        const unsigned int key = mapkey(row[i]);
        if (round == 0 || (key >> (shift + 8)) == prefix)
          atomicAdd(&hist[(key >> shift) & 255u], 1u);
      }
      __syncthreads();
      if (t == 0) {
        unsigned int need = s_needed, cum = 0u; int bb = 0;
        for (int bin = 255; bin >= 0; --bin) {
          const unsigned int h = hist[bin];
          if (cum + h >= need) { bb = bin; break; }
          cum += h;
        }
        s_prefix = (s_prefix << 8) | (unsigned int)bb;
        s_needed = need - cum;
      }
      __syncthreads();
    }

    const unsigned int T = s_prefix;
    const int need_eq = (int)s_needed;  // how many key==T to take
    const int base_eq = K - need_eq;    // count of key>T (all taken)

    // Deterministic compaction: per-thread counts -> serial exclusive scan -> write
    unsigned int myGt = 0, myEq = 0;
    for (int i = t; i < F_DIM; i += 256) {
      if (masked && !((deadmask[i >> 5] >> mysh) & 1u)) continue;
      const unsigned int key = mapkey(row[i]);
      myGt += (key > T);
      myEq += (key == T);
    }
    cntGt[t] = myGt; cntEq[t] = myEq;
    for (int i = t; i < K; i += 256) { selF[i] = 0; selW[i] = 0.f; }  // safety init
    __syncthreads();
    if (t == 0) {
      unsigned int rg = 0, re = 0;
      for (int j = 0; j < 256; ++j) {
        const unsigned int g = cntGt[j]; cntGt[j] = rg; rg += g;
        const unsigned int e = cntEq[j]; cntEq[j] = re; re += e;
      }
    }
    __syncthreads();
    int pg = (int)cntGt[t];
    int pe = (int)cntEq[t];
    for (int i = t; i < F_DIM; i += 256) {
      if (masked && !((deadmask[i >> 5] >> mysh) & 1u)) continue;
      const float x = row[i];
      const unsigned int key = mapkey(x);
      if (key > T) {
        if (pg < base_eq) { selF[pg] = i; selW[pg] = x; }
        ++pg;
      } else if (key == T) {
        if (pe < need_eq) { selF[base_eq + pe] = i; selW[base_eq + pe] = x; }
        ++pe;
      }
    }
    __syncthreads();

    // Reconstruction: out[e] = sum_k w_k * lookup[f_k, e] (+ bias for pass 0)
    const int e0 = t * 4;  // 256 threads x float4 = 1024 = E
    float4 acc = make_float4(0.f, 0.f, 0.f, 0.f);
    for (int k2 = 0; k2 < K; ++k2) {
      const float w = selW[k2];
      const int   f = selF[k2];
      const float4 v = *(const float4*)&lookup[(size_t)f * E_DIM + e0];
      acc.x += w * v.x; acc.y += w * v.y; acc.z += w * v.z; acc.w += w * v.w;
    }
    size_t obase;
    if (pass == 0) {
      const float4 bb = *(const float4*)&enc_bias[e0];
      acc.x += bb.x; acc.y += bb.y; acc.z += bb.z; acc.w += bb.w;
      obase = (size_t)b * E_DIM;
    } else {
      obase = (size_t)B_DIM * E_DIM + (size_t)b * E_DIM;
    }
    *(float4*)&out[obase + e0] = acc;
    __syncthreads();  // before shared reuse in next pass
  }
}

// ---------------------------------------------------------------------------
extern "C" void kernel_launch(void* const* d_in, const int* in_sizes, int n_in,
                              void* d_out, int out_size, void* d_ws, size_t ws_size,
                              hipStream_t stream) {
  (void)in_sizes; (void)n_in; (void)out_size;
  const float* embed      = (const float*)d_in[0];
  const float* enc_bias   = (const float*)d_in[1];
  const float* enc_W      = (const float*)d_in[2];
  const float* lookup     = (const float*)d_in[3];
  const int*   last_usage = (const int*)d_in[4];
  float* out  = (float*)d_out;
  float* proj = (float*)d_ws;

  // Row chunk: fit projection chunk in workspace; <=512 rows (64 MB) keeps it L2-resident.
  const size_t rowBytes = (size_t)F_DIM * sizeof(float);
  int chunk = 512;
  while (chunk > 64 && (size_t)chunk * rowBytes > ws_size) chunk >>= 1;

  for (int r0 = 0; r0 < B_DIM; r0 += chunk) {
    dim3 g1(chunk / 64, F_DIM / 64);
    gemm_proj_kernel<<<g1, 256, 0, stream>>>(embed, enc_bias, enc_W, proj, r0);
    select_recon_kernel<<<chunk, 256, 0, stream>>>(proj, lookup, enc_bias, last_usage, out, r0);
  }
}
